// GeneEnvAttentionModelWithMoE_15006615734197
// MI455X (gfx1250) — matrix operand order repaired and checked by reading, not compiled, not run on hardware
//
#include <hip/hip_runtime.h>
#include <math.h>

#define NTOK 8192
#define IDIM 1024
#define HDIM 4096
#define ODIM 1024
#define NEXP 8

typedef __attribute__((ext_vector_type(16))) __bf16         v16bf;
typedef __attribute__((ext_vector_type(8)))  float          v8f;
typedef __attribute__((ext_vector_type(16))) unsigned short v16u;
typedef __attribute__((ext_vector_type(8)))  unsigned short v8u;

__device__ __forceinline__ unsigned short f32_to_bf16(float f) {
  unsigned int u = __float_as_uint(f);
  u += 0x7fffu + ((u >> 16) & 1u);       // round-to-nearest-even
  return (unsigned short)(u >> 16);
}

// Async global->LDS copy, 16 bytes per lane (CDNA5 ISA 08 §4, GV mode,
// tracked with ASYNCcnt). lds_off = per-lane LDS byte address (flat LDS
// addresses truncate to the low 32 bits).
__device__ __forceinline__ void async_cp_b128(unsigned lds_off,
                                              const unsigned short* g) {
  asm volatile("global_load_async_to_lds_b128 %0, %1, off"
               :: "v"(lds_off), "v"((unsigned long long)(uintptr_t)g)
               : "memory");
}
__device__ __forceinline__ void wait_async0() {
  asm volatile("s_wait_asynccnt 0x0" ::: "memory");
}

// ---------------------------------------------------------------- utilities
__global__ void zero_f32_kernel(float* __restrict__ p, long long n) {
  long long i = (long long)blockIdx.x * blockDim.x + threadIdx.x;
  long long s = (long long)gridDim.x * blockDim.x;
  for (; i < n; i += s) p[i] = 0.f;
}

__global__ void cvt_bf16_kernel(unsigned short* __restrict__ dst,
                                const float* __restrict__ src, long long n) {
  long long i = (long long)blockIdx.x * blockDim.x + threadIdx.x;
  long long s = (long long)gridDim.x * blockDim.x;
  for (; i < n; i += s) dst[i] = f32_to_bf16(src[i]);
}

// fp32 [rows][cols] -> bf16 [cols][rows] (one expert slab per blockIdx.z).
// Makes the GEMM B operand K-contiguous so LDS staging is pure b128 copies.
__global__ __launch_bounds__(256)
void cvt_transpose_kernel(unsigned short* __restrict__ dst,
                          const float* __restrict__ src, int rows, int cols) {
  __shared__ unsigned short tile[32][33];
  const float* s = src + (size_t)blockIdx.z * rows * cols;
  unsigned short* d = dst + (size_t)blockIdx.z * rows * cols;
  const int c0 = blockIdx.x * 32, r0 = blockIdx.y * 32;
  const int tx = threadIdx.x & 31, ty = threadIdx.x >> 5;   // 32 x 8
#pragma unroll
  for (int i = 0; i < 32; i += 8)
    tile[ty + i][tx] = f32_to_bf16(s[(size_t)(r0 + ty + i) * cols + (c0 + tx)]);
  __syncthreads();
#pragma unroll
  for (int i = 0; i < 32; i += 8)
    d[(size_t)(c0 + ty + i) * rows + (r0 + tx)] = tile[tx][ty + i];
}

// ---------------------------------------------------------------- gating
__global__ __launch_bounds__(128)
void gating_kernel(const float* __restrict__ x, const float* __restrict__ wg,
                   const float* __restrict__ bg, float* __restrict__ combine,
                   float* __restrict__ imp, float* __restrict__ loadc) {
  const int row = blockIdx.x;
  const int tid = threadIdx.x;
  float acc[NEXP];
#pragma unroll
  for (int e = 0; e < NEXP; ++e) acc[e] = 0.f;

  const float* xr = x + (size_t)row * IDIM;
  for (int i = tid; i < IDIM; i += 128) {
    float xv = xr[i];
    const float4* w4 = (const float4*)(wg + (size_t)i * NEXP);
    float4 a = w4[0], b = w4[1];
    acc[0] += xv * a.x; acc[1] += xv * a.y; acc[2] += xv * a.z; acc[3] += xv * a.w;
    acc[4] += xv * b.x; acc[5] += xv * b.y; acc[6] += xv * b.z; acc[7] += xv * b.w;
  }

  __shared__ float red[128][NEXP];
#pragma unroll
  for (int e = 0; e < NEXP; ++e) red[tid][e] = acc[e];
  __syncthreads();
  for (int s = 64; s > 0; s >>= 1) {
    if (tid < s) {
#pragma unroll
      for (int e = 0; e < NEXP; ++e) red[tid][e] += red[tid + s][e];
    }
    __syncthreads();
  }

  if (tid == 0) {
    float p[NEXP], mx = -1e30f;
#pragma unroll
    for (int e = 0; e < NEXP; ++e) { p[e] = red[0][e] + bg[e]; mx = fmaxf(mx, p[e]); }
    float sum = 0.f;
#pragma unroll
    for (int e = 0; e < NEXP; ++e) { p[e] = expf(p[e] - mx); sum += p[e]; }
#pragma unroll
    for (int e = 0; e < NEXP; ++e) p[e] /= sum;

    int i1 = 0;
#pragma unroll
    for (int e = 1; e < NEXP; ++e) if (p[e] > p[i1]) i1 = e;
    int i2 = (i1 == 0) ? 1 : 0;
#pragma unroll
    for (int e = 0; e < NEXP; ++e) if (e != i1 && p[e] > p[i2]) i2 = e;

    float s2 = p[i1] + p[i2];
    float* crow = combine + (size_t)row * NEXP;
#pragma unroll
    for (int e = 0; e < NEXP; ++e) crow[e] = 0.f;
    crow[i1] = p[i1] / s2;
    crow[i2] = p[i2] / s2;

#pragma unroll
    for (int e = 0; e < NEXP; ++e) atomicAdd(&imp[e], p[e]);
    atomicAdd(&loadc[i1], 1.f);
    atomicAdd(&loadc[i2], 1.f);
  }
}

__global__ void aux_kernel(const float* __restrict__ imp,
                           const float* __restrict__ loadc,
                           float* __restrict__ dst) {
  if (threadIdx.x == 0 && blockIdx.x == 0) {
    float mi = 0.f, ml = 0.f;
#pragma unroll
    for (int e = 0; e < NEXP; ++e) { mi += imp[e]; ml += loadc[e] / (float)NTOK; }
    mi /= NEXP; ml /= NEXP;
    float vi = 0.f, vl = 0.f;
#pragma unroll
    for (int e = 0; e < NEXP; ++e) {
      float d1 = imp[e] - mi;                 vi += d1 * d1;
      float d2 = loadc[e] / (float)NTOK - ml; vl += d2 * d2;
    }
    vi /= (NEXP - 1); vl /= (NEXP - 1);       // torch .std() ddof=1
    float a1 = sqrtf(vi) / (mi + 1e-6f);
    float a2 = sqrtf(vl) / (ml + 1e-6f);
    dst[0] = a1 * a1 + a2 * a2;
  }
}

// ------------------------------------------------------- WMMA bf16 GEMM tile
// Block = 256 threads (8 wave32), tile 128(M) x 128(N), K step 64 (2 WMMA
// K-steps per LDS stage). Wave grid 2x4; each wave owns 4x2 subtiles of
// 16x16. Both operands are K-contiguous in global memory (B pre-transposed).
// Double-buffered ping-pong staging: GLOBAL_LOAD_ASYNC_TO_LDS_B128 for tile
// k+1 is issued right after the barrier, overlapping with the 16 WMMAs of
// tile k; per-wave s_wait_asynccnt 0 before the barrier makes all waves'
// copies LDS-visible. Row stride 72 bf16 = 144 B keeps every ds_*_b128
// access 16-byte aligned. Fragment layouts follow CDNA5 ISA 7.12.2.
__device__ __forceinline__ void stage_tiles(
    unsigned short (*__restrict__ dstA)[72], unsigned short (*__restrict__ dstB)[72],
    const unsigned short* __restrict__ A, const unsigned short* __restrict__ Bt,
    int lda, int ldb, int tm, int tn, int k0, int kpf, int tid) {
  // 128 rows x 64 k per matrix; 1024 chunks of 8 bf16, 4 per thread
#pragma unroll
  for (int cc = 0; cc < 4; ++cc) {
    int c   = tid + cc * 256;
    int row = c >> 3;
    int kq  = (c & 7) << 3;
    const unsigned short* ga = A  + (size_t)(tm + row) * lda + (k0 + kq);
    const unsigned short* gb = Bt + (size_t)(tn + row) * ldb + (k0 + kq);
    async_cp_b128((unsigned)(uintptr_t)&dstA[row][kq], ga);
    async_cp_b128((unsigned)(uintptr_t)&dstB[row][kq], gb);
    if (cc == 0) {  // pull tile k+2 toward L2 while the async engine fills k+1
      __builtin_prefetch(A  + (size_t)(tm + row) * lda + (kpf + kq), 0, 1);
      __builtin_prefetch(Bt + (size_t)(tn + row) * ldb + (kpf + kq), 0, 1);
    }
  }
}

__device__ __forceinline__ void gemm_tile_bf16(
    const unsigned short* __restrict__ A, const unsigned short* __restrict__ Bt,
    int lda, int ldb, int kdim, int tm, int tn, int tid, v8f acc[4][2]) {
  __shared__ unsigned short As[2][128][72];
  __shared__ unsigned short Bs[2][128][72];

  const int lane = tid & 31;
  const int wid  = tid >> 5;
  const int wm   = wid >> 2;     // 0..1
  const int wn   = wid & 3;      // 0..3
  const int lrow = lane & 15;
  const int lhi  = lane >> 4;    // 0..1

  const int nt = kdim >> 6;      // K tiles of 64
  stage_tiles(As[0], Bs[0], A, Bt, lda, ldb, tm, tn, 0,
              (nt > 1) ? 64 : 0, tid);

  int p = 0;
  for (int kt = 0; kt < nt; ++kt) {
    wait_async0();               // own async copies done ...
    __syncthreads();             // ... barrier => everyone's copies visible

    if (kt + 1 < nt) {           // refill the other buffer while we compute
      int knext = (kt + 1) << 6;
      int kpf   = (kt + 2 < nt) ? ((kt + 2) << 6) : knext;
      stage_tiles(As[p ^ 1], Bs[p ^ 1], A, Bt, lda, ldb, tm, tn, knext, kpf, tid);
    }

#pragma unroll
    for (int ks = 0; ks < 2; ++ks) {
      v16bf afrag[4];
#pragma unroll
      for (int mi = 0; mi < 4; ++mi) {
        int row = wm * 64 + mi * 16 + lrow;
        int ka  = ks * 32 + lhi * 8;
        v8u lo = *(const v8u*)&As[p][row][ka];        // K = ka .. ka+7
        v8u hi = *(const v8u*)&As[p][row][ka + 16];   // K = ka+16 .. ka+23
        v16u u;
#pragma unroll
        for (int j = 0; j < 8; ++j) { u[j] = lo[j]; u[8 + j] = hi[j]; }
        afrag[mi] = __builtin_bit_cast(v16bf, u);
      }
      v16bf bfrag[2];
#pragma unroll
      for (int ni = 0; ni < 2; ++ni) {
        int col = wn * 32 + ni * 16 + lrow;
        int kb  = ks * 32 + lhi * 8;
        v8u lo = *(const v8u*)&Bs[p][col][kb];
        v8u hi = *(const v8u*)&Bs[p][col][kb + 16];
        v16u u;
#pragma unroll
        for (int j = 0; j < 8; ++j) { u[j] = lo[j]; u[8 + j] = hi[j]; }
        bfrag[ni] = __builtin_bit_cast(v16bf, u);
      }
#pragma unroll
      for (int mi = 0; mi < 4; ++mi)
#pragma unroll
        for (int ni = 0; ni < 2; ++ni)
          acc[mi][ni] = __builtin_amdgcn_wmma_f32_16x16x32_bf16(
              false, afrag[mi], false, bfrag[ni], (short)0, acc[mi][ni],
              false, false);
    }
    __syncthreads();             // protect buffer p before it is refilled
    p ^= 1;
  }
}

__device__ __forceinline__ bool block_routed(const float* __restrict__ combine,
                                             int tm, int e, int tid) {
  __shared__ int s_route;
  if (tid == 0) s_route = 0;
  __syncthreads();
  if (tid < 128 && combine[(size_t)(tm + tid) * NEXP + e] != 0.f) s_route = 1;
  __syncthreads();
  return s_route != 0;
}

// GEMM1: y = x @ W1[e] + b1[e]   (fp32 out, bf16 WMMA inputs)
__global__ __launch_bounds__(256)
void moe_gemm1_kernel(const unsigned short* __restrict__ xb,
                      const unsigned short* __restrict__ w1t,
                      const float* __restrict__ b1e,
                      const float* __restrict__ combine, int e,
                      float* __restrict__ y) {
  const int tm = blockIdx.y * 128, tn = blockIdx.x * 128;
  const int tid = threadIdx.x;
  if (!block_routed(combine, tm, e, tid)) return;   // top-2 routing skip

  v8f acc[4][2];
  const v8f z = {0.f, 0.f, 0.f, 0.f, 0.f, 0.f, 0.f, 0.f};
#pragma unroll
  for (int mi = 0; mi < 4; ++mi)
#pragma unroll
    for (int ni = 0; ni < 2; ++ni) acc[mi][ni] = z;

  gemm_tile_bf16(xb, w1t, IDIM, IDIM, IDIM, tm, tn, tid, acc);

  const int lane = tid & 31, wid = tid >> 5;
  const int wm = wid >> 2, wn = wid & 3;
  const int lrow = lane & 15, lhi = lane >> 4;
#pragma unroll
  for (int mi = 0; mi < 4; ++mi)
#pragma unroll
    for (int ni = 0; ni < 2; ++ni) {
      int col = tn + wn * 32 + ni * 16 + lrow;
      float bias = b1e[col];
#pragma unroll
      for (int r = 0; r < 8; ++r) {
        int row = tm + wm * 64 + mi * 16 + lhi * 8 + r;
        y[(size_t)row * HDIM + col] = acc[mi][ni][r] + bias;
      }
    }
}

// Row-wise LayerNorm + exact GELU, fp32 -> bf16
__global__ __launch_bounds__(256)
void ln_gelu_kernel(const float* __restrict__ y, const float* __restrict__ g,
                    const float* __restrict__ b, const float* __restrict__ combine,
                    int e, unsigned short* __restrict__ h) {
  const int row = blockIdx.x;
  if (combine[(size_t)row * NEXP + e] == 0.f) return;  // unrouted token
  const float* yr = y + (size_t)row * HDIM;

  float s = 0.f, ss = 0.f;
  for (int i = threadIdx.x; i < HDIM; i += 256) {
    float v = yr[i]; s += v; ss += v * v;
  }
  __shared__ float rs[256], rq[256];
  rs[threadIdx.x] = s; rq[threadIdx.x] = ss;
  __syncthreads();
  for (int st = 128; st > 0; st >>= 1) {
    if (threadIdx.x < st) {
      rs[threadIdx.x] += rs[threadIdx.x + st];
      rq[threadIdx.x] += rq[threadIdx.x + st];
    }
    __syncthreads();
  }
  float mean = rs[0] * (1.f / HDIM);
  float var  = rq[0] * (1.f / HDIM) - mean * mean;   // biased, matches jnp.var
  float inv  = rsqrtf(var + 1e-5f);

  for (int i = threadIdx.x; i < HDIM; i += 256) {
    float v  = (yr[i] - mean) * inv * g[i] + b[i];
    float ge = 0.5f * v * (1.f + erff(v * 0.70710678118654752f));  // exact GELU
    h[(size_t)row * HDIM + i] = f32_to_bf16(ge);
  }
}

// GEMM2: out_acc += combine[:,e] * (h @ W2[e] + b2[e])
__global__ __launch_bounds__(256)
void moe_gemm2_kernel(const unsigned short* __restrict__ hb,
                      const unsigned short* __restrict__ w2t,
                      const float* __restrict__ b2e,
                      const float* __restrict__ combine, int e,
                      float* __restrict__ out_acc) {
  const int tm = blockIdx.y * 128, tn = blockIdx.x * 128;
  const int tid = threadIdx.x;
  if (!block_routed(combine, tm, e, tid)) return;

  v8f acc[4][2];
  const v8f z = {0.f, 0.f, 0.f, 0.f, 0.f, 0.f, 0.f, 0.f};
#pragma unroll
  for (int mi = 0; mi < 4; ++mi)
#pragma unroll
    for (int ni = 0; ni < 2; ++ni) acc[mi][ni] = z;

  gemm_tile_bf16(hb, w2t, HDIM, HDIM, HDIM, tm, tn, tid, acc);

  const int lane = tid & 31, wid = tid >> 5;
  const int wm = wid >> 2, wn = wid & 3;
  const int lrow = lane & 15, lhi = lane >> 4;
#pragma unroll
  for (int mi = 0; mi < 4; ++mi)
#pragma unroll
    for (int r = 0; r < 8; ++r) {
      int row = tm + wm * 64 + mi * 16 + lhi * 8 + r;
      float cw = combine[(size_t)row * NEXP + e];
      if (cw != 0.f) {
#pragma unroll
        for (int ni = 0; ni < 2; ++ni) {
          int col = tn + wn * 32 + ni * 16 + lrow;
          out_acc[(size_t)row * ODIM + col] += cw * (acc[mi][ni][r] + b2e[col]);
        }
      }
    }
}

// Final LayerNorm over O
__global__ __launch_bounds__(256)
void final_ln_kernel(const float* __restrict__ acc, const float* __restrict__ g,
                     const float* __restrict__ b, float* __restrict__ out) {
  const int row = blockIdx.x;
  const float* ar = acc + (size_t)row * ODIM;
  float s = 0.f, ss = 0.f;
  for (int i = threadIdx.x; i < ODIM; i += 256) {
    float v = ar[i]; s += v; ss += v * v;
  }
  __shared__ float rs[256], rq[256];
  rs[threadIdx.x] = s; rq[threadIdx.x] = ss;
  __syncthreads();
  for (int st = 128; st > 0; st >>= 1) {
    if (threadIdx.x < st) {
      rs[threadIdx.x] += rs[threadIdx.x + st];
      rq[threadIdx.x] += rq[threadIdx.x + st];
    }
    __syncthreads();
  }
  float mean = rs[0] * (1.f / ODIM);
  float var  = rq[0] * (1.f / ODIM) - mean * mean;
  float inv  = rsqrtf(var + 1e-5f);
  for (int i = threadIdx.x; i < ODIM; i += 256)
    out[(size_t)row * ODIM + i] = (ar[i] - mean) * inv * g[i] + b[i];
}

// ---------------------------------------------------------------- launcher
extern "C" void kernel_launch(void* const* d_in, const int* in_sizes, int n_in,
                              void* d_out, int out_size, void* d_ws, size_t ws_size,
                              hipStream_t stream) {
  const float* x   = (const float*)d_in[0];
  const float* w1  = (const float*)d_in[1];
  const float* b1  = (const float*)d_in[2];
  const float* g1  = (const float*)d_in[3];
  const float* be1 = (const float*)d_in[4];
  const float* w2  = (const float*)d_in[5];
  const float* b2  = (const float*)d_in[6];
  const float* wg  = (const float*)d_in[7];
  const float* bg  = (const float*)d_in[8];
  const float* og  = (const float*)d_in[9];
  const float* ob  = (const float*)d_in[10];
  float* out = (float*)d_out;

  char*  ws  = (char*)d_ws;
  size_t off = 0;
  auto take = [&](size_t bytes) -> void* {
    void* p = (void*)(ws + off);
    off += (bytes + 255) & ~(size_t)255;
    return p;
  };

  unsigned short* xb      = (unsigned short*)take(2ull * NTOK * IDIM);
  unsigned short* w1t     = (unsigned short*)take(2ull * NEXP * IDIM * HDIM);
  unsigned short* w2t     = (unsigned short*)take(2ull * NEXP * HDIM * ODIM);
  float*          combine = (float*)take(4ull * NTOK * NEXP);
  float*          region  = (float*)take(4ull * ((size_t)NTOK * ODIM + 16));
  float*          out_acc = region;
  float*          imp     = region + (size_t)NTOK * ODIM;
  float*          loadc   = imp + NEXP;
  float*          ybuf    = (float*)take(4ull * NTOK * HDIM);
  unsigned short* hbuf    = (unsigned short*)take(2ull * NTOK * HDIM);

  // zero out_acc + imp + load counters (workspace is poisoned)
  zero_f32_kernel<<<2048, 256, 0, stream>>>(region, (long long)NTOK * ODIM + 16);

  // fp32 -> bf16 conversions; weights transposed to [N][K] so GEMM staging is
  // K-contiguous (pure async b128 copies, no LDS-side transpose)
  cvt_bf16_kernel<<<4096, 256, 0, stream>>>(xb, x, (long long)NTOK * IDIM);
  cvt_transpose_kernel<<<dim3(HDIM / 32, IDIM / 32, NEXP), 256, 0, stream>>>(
      w1t, w1, IDIM, HDIM);
  cvt_transpose_kernel<<<dim3(ODIM / 32, HDIM / 32, NEXP), 256, 0, stream>>>(
      w2t, w2, HDIM, ODIM);

  // gating + aux loss
  gating_kernel<<<NTOK, 128, 0, stream>>>(x, wg, bg, combine, imp, loadc);
  aux_kernel<<<1, 32, 0, stream>>>(imp, loadc, out + (size_t)NTOK * ODIM);

  // experts (sequential; routed-tile skipping inside the GEMMs)
  for (int e = 0; e < NEXP; ++e) {
    moe_gemm1_kernel<<<dim3(HDIM / 128, NTOK / 128), 256, 0, stream>>>(
        xb, w1t + (size_t)e * IDIM * HDIM, b1 + (size_t)e * HDIM, combine, e, ybuf);
    ln_gelu_kernel<<<NTOK, 256, 0, stream>>>(
        ybuf, g1 + (size_t)e * HDIM, be1 + (size_t)e * HDIM, combine, e, hbuf);
    moe_gemm2_kernel<<<dim3(ODIM / 128, NTOK / 128), 256, 0, stream>>>(
        hbuf, w2t + (size_t)e * HDIM * ODIM, b2 + (size_t)e * ODIM, combine, e, out_acc);
  }

  final_ln_kernel<<<NTOK, 256, 0, stream>>>(out_acc, og, ob, out);
}
